// TSMambaBlock_78099685310819
// MI455X (gfx1250) — matrix operand order repaired and checked by reading, not compiled, not run on hardware
//
#include <hip/hip_runtime.h>
#include <hip/hip_bf16.h>

// ---------------------------------------------------------------------------
// TSMamba block for MI455X (gfx1250, wave32, WMMA 16x16x32 f16)
// ---------------------------------------------------------------------------

typedef _Float16 half_t;
typedef __attribute__((ext_vector_type(16))) _Float16 v16h;
typedef __attribute__((ext_vector_type(8)))  _Float16 v8h;
typedef __attribute__((ext_vector_type(8)))  float    v8f;

#define EMB     192
#define DIN     384       // d_inner
#define DIN2    768       // 2*d_inner
#define DSTATE  16
#define DTRANK  12
#define DCONV   4
#define NPEDS   2347
#define NP_PAD  2352      // ceil16(2347)
#define TSEQ    12        // 11 previous + current
#define MT      (NPEDS*TSEQ)      // 28164 temporal tokens
#define MT_PAD  28176             // ceil16
#define NSCENES 64
#define MAXP    58
#define MS      (NSCENES*MAXP)    // 3712, multiple of 16
#define XPN     48        // x_proj rows 44 padded to 48
#define KDT     32        // dt_rank 12 padded to 32

// ---------------------------------------------------------------------------
// Wave-level WMMA GEMM:  C[M,N] = A[M,K] * W[N,K]^T  (f16 in, f32 out)
// Each wave computes a 16 x (16*NT) tile: the A fragment is loaded once per
// K-step and reused across NT B fragments / accumulators (4x less A traffic,
// 4x denser WMMA issue vs one-tile-per-wave).
// grid = (N/(16*NT), Mpad/16), block = 32 (one wave)
// A fragment (16-bit A 16x32): lane&15 = M row; lane>=16 selects K 16..31
// B fragment (16-bit B 32x16): lane&15 = N col; lane>=16 selects K 16..31
// C/D: 8 VGPRs; VGPR r -> M = r (+8 for upper lanes), N = lane&15
// ---------------------------------------------------------------------------
template <int NT>
__global__ __launch_bounds__(32)
void gemm_wmma_f16(const half_t* __restrict__ A, const half_t* __restrict__ W,
                   float* __restrict__ C, int N, int K) {
  const int lane = threadIdx.x;
  const int r    = lane & 15;
  const int hi   = lane >> 4;
  const size_t m0 = (size_t)blockIdx.y * 16;
  const size_t n0 = (size_t)blockIdx.x * (16 * NT);

  const half_t* ap = A + (m0 + (size_t)r) * (size_t)K + hi * 16;
  const half_t* bp = W + (n0 + (size_t)r) * (size_t)K + hi * 16;

  v8f acc[NT];
  #pragma unroll
  for (int t = 0; t < NT; ++t) acc[t] = (v8f){};

  for (int k0 = 0; k0 < K; k0 += 32) {
    v8h a0 = *(const v8h*)(ap + k0);
    v8h a1 = *(const v8h*)(ap + k0 + 8);
    v16h a;
    #pragma unroll
    for (int i = 0; i < 8; ++i) { a[i] = a0[i]; a[i + 8] = a1[i]; }
    #pragma unroll
    for (int t = 0; t < NT; ++t) {
      v16h b = *(const v16h*)(bp + (size_t)t * 16 * K + k0);
      acc[t] = __builtin_amdgcn_wmma_f32_16x16x32_f16(
          /*neg_a=*/false, a, /*neg_b=*/false, b,
          /*c_mod=*/(short)0, acc[t], /*reuse_a=*/false, /*reuse_b=*/false);
    }
  }

  #pragma unroll
  for (int t = 0; t < NT; ++t) {
    float* cp = C + (m0 + (size_t)(hi * 8)) * (size_t)N + n0 + t * 16 + r;
    #pragma unroll
    for (int i = 0; i < 8; ++i) cp[(size_t)i * N] = acc[t][i];
  }
}

// ---------------------------------------------------------------------------
// Weight convert fp32 -> f16 with zero padding:  dst[Npad,Kpad] <- src[N,K]
// ---------------------------------------------------------------------------
__global__ void k_convert_w(half_t* __restrict__ dst, const float* __restrict__ src,
                            int N, int K, int Npad, int Kpad) {
  long i = (long)blockIdx.x * blockDim.x + threadIdx.x;
  long tot = (long)Npad * Kpad;
  if (i >= tot) return;
  int k = (int)(i % Kpad);
  int n = (int)(i / Kpad);
  dst[i] = (n < N && k < K) ? (half_t)src[(long)n * K + k] : (half_t)0.f;
}

__global__ void k_prep_A(float* __restrict__ dst, const float* __restrict__ alog) {
  int i = blockIdx.x * blockDim.x + threadIdx.x;
  if (i < DIN * DSTATE) dst[i] = -__expf(alog[i]);
}

// ---------------------------------------------------------------------------
// LayerNorm over EMB=192, one block (256 thr) per pedestrian
// ---------------------------------------------------------------------------
__global__ __launch_bounds__(256)
void k_layernorm(float* __restrict__ xn, const float* __restrict__ x,
                 const float* __restrict__ g, const float* __restrict__ b) {
  __shared__ float s1[256], s2[256];
  const int ped = blockIdx.x;
  const int t   = threadIdx.x;
  float v = (t < EMB) ? x[(long)ped * EMB + t] : 0.f;
  s1[t] = v; s2[t] = v * v;
  __syncthreads();
  #pragma unroll
  for (int s = 128; s > 0; s >>= 1) {
    if (t < s) { s1[t] += s1[t + s]; s2[t] += s2[t + s]; }
    __syncthreads();
  }
  const float mu  = s1[0] * (1.f / EMB);
  const float var = s2[0] * (1.f / EMB) - mu * mu;
  if (t < EMB)
    xn[(long)ped * EMB + t] = (v - mu) * rsqrtf(var + 1e-5f) * g[t] + b[t];
}

// combined[ped*12+t, e] = t<11 ? prev[t,ped,e] : xn[ped,e]   (as f16)
__global__ void k_build_combined(half_t* __restrict__ dst,
                                 const float* __restrict__ prev,
                                 const float* __restrict__ xn) {
  long i = (long)blockIdx.x * blockDim.x + threadIdx.x;
  if (i >= (long)MT * EMB) return;
  int  e   = (int)(i % EMB);
  long tok = i / EMB;
  int  t   = (int)(tok % TSEQ);
  long ped = tok / TSEQ;
  float v = (t < TSEQ - 1) ? prev[((long)t * NPEDS + ped) * EMB + e]
                           : xn[ped * EMB + e];
  dst[i] = (half_t)v;
}

// ---------------------------------------------------------------------------
// Causal depthwise conv (width 4) + bias + SiLU.  xz row stride = 768,
// conv over the first 384 columns.  Writes f32 (for scan) and f16 (for GEMM).
// ---------------------------------------------------------------------------
__global__ void k_conv_silu(float* __restrict__ xc_f, half_t* __restrict__ xc_h,
                            const float* __restrict__ xz,
                            const float* __restrict__ cw, const float* __restrict__ cb,
                            int B, int L) {
  long i = (long)blockIdx.x * blockDim.x + threadIdx.x;
  long tot = (long)B * L * DIN;
  if (i >= tot) return;
  int  e   = (int)(i % DIN);
  long tok = i / DIN;
  int  t   = (int)(tok % L);
  long b   = tok / L;
  float s = cb[e];
  #pragma unroll
  for (int k = 0; k < DCONV; ++k) {
    int j = t - (DCONV - 1) + k;
    if (j >= 0) s += cw[e * DCONV + k] * xz[((long)b * L + j) * DIN2 + e];
  }
  float o = s * (1.f / (1.f + __expf(-s)));
  xc_f[tok * DIN + e] = o;
  xc_h[tok * DIN + e] = (half_t)o;
}

// dtin[tok, 0..31] = k<12 ? xdbl[tok, k] : 0   (f16, K padded for WMMA)
__global__ void k_prep_dt(half_t* __restrict__ dst, const float* __restrict__ xdbl,
                          long Mtok) {
  long i = (long)blockIdx.x * blockDim.x + threadIdx.x;
  if (i >= Mtok * KDT) return;
  int  k   = (int)(i % KDT);
  long tok = i / KDT;
  dst[i] = (k < DTRANK) ? (half_t)xdbl[tok * XPN + k] : (half_t)0.f;
}

// ---------------------------------------------------------------------------
// Selective scan.  One thread per (batch, channel); 16-wide state in regs.
// dt = softplus(dtraw + bias);  h = h*exp(dt*A) + dt*x*B;  y = <h,C>
// out = (y + x*D) * silu(z)   -> f16 rows for out_proj GEMM
// last_only=1: emit only t=L-1 (temporal path, B rows out)
// ---------------------------------------------------------------------------
__global__ void k_scan(half_t* __restrict__ yout,
                       const float* __restrict__ xz,    // z at cols 384..767
                       const float* __restrict__ xc,
                       const float* __restrict__ xdbl,  // [.,48]: dt|B|C
                       const float* __restrict__ dtraw, // [.,384]
                       const float* __restrict__ Aneg,  // [384,16]
                       const float* __restrict__ Dv, const float* __restrict__ dtb,
                       int B, int L, int last_only) {
  long i = (long)blockIdx.x * blockDim.x + threadIdx.x;
  if (i >= (long)B * DIN) return;
  const int e = (int)(i % DIN);
  const int b = (int)(i / DIN);

  float h[DSTATE];
  float Ar[DSTATE];
  #pragma unroll
  for (int s = 0; s < DSTATE; ++s) { h[s] = 0.f; Ar[s] = Aneg[e * DSTATE + s]; }
  const float bias = dtb[e];
  const float De   = Dv[e];

  for (int t = 0; t < L; ++t) {
    const long tok = (long)b * L + t;
    float dt = dtraw[tok * DIN + e] + bias;
    dt = (dt > 20.f) ? dt : log1pf(__expf(dt));       // softplus
    const float xt = xc[tok * DIN + e];
    const float dx = dt * xt;
    const float* xd = xdbl + tok * XPN;
    float y = 0.f;
    #pragma unroll
    for (int s = 0; s < DSTATE; ++s) {
      h[s] = h[s] * __expf(dt * Ar[s]) + dx * xd[DTRANK + s];
      y += h[s] * xd[DTRANK + DSTATE + s];
    }
    if (!last_only || t == L - 1) {
      const float yf = y + xt * De;
      const float z  = xz[tok * DIN2 + DIN + e];
      const float o  = yf * (z * (1.f / (1.f + __expf(-z))));
      const long  oi = last_only ? ((long)b * DIN + e) : (tok * DIN + e);
      yout[oi] = (half_t)o;
    }
  }
}

// exclusive prefix sum of 64 scene lengths (single lane, trivial size)
__global__ void k_offsets(int* __restrict__ off, const int* __restrict__ lens) {
  if (blockIdx.x == 0 && threadIdx.x == 0) {
    int a = 0;
    for (int s = 0; s < NSCENES; ++s) { off[s] = a; a += lens[s]; }
  }
}

// gathered[slot, e] = valid ? xn[ped, e] : 0   (f16, padded scene layout)
__global__ void k_gather(half_t* __restrict__ dst, const float* __restrict__ xn,
                         const int* __restrict__ off, const int* __restrict__ lens) {
  long i = (long)blockIdx.x * blockDim.x + threadIdx.x;
  if (i >= (long)MS * EMB) return;
  int e    = (int)(i % EMB);
  int slot = (int)(i / EMB);
  int s = slot / MAXP, j = slot % MAXP;
  float v = 0.f;
  if (j < lens[s]) v = xn[((long)off[s] + j) * EMB + e];
  dst[i] = (half_t)v;
}

// out[ped] = xn[ped] + temp[ped] + spa[slot(ped)]
__global__ void k_final(float* __restrict__ out, const float* __restrict__ xn,
                        const float* __restrict__ temp, const float* __restrict__ spa,
                        const int* __restrict__ off, const int* __restrict__ lens) {
  long i = (long)blockIdx.x * blockDim.x + threadIdx.x;
  if (i >= (long)MS * EMB) return;
  int e    = (int)(i % EMB);
  int slot = (int)(i / EMB);
  int s = slot / MAXP, j = slot % MAXP;
  if (j < lens[s]) {
    long ped = (long)off[s] + j;
    out[ped * EMB + e] = xn[ped * EMB + e] + temp[ped * EMB + e]
                       + spa[(long)slot * EMB + e];
  }
}

// ---------------------------------------------------------------------------
// Host side
// ---------------------------------------------------------------------------
static inline char* bump(char*& p, size_t bytes) {
  char* r = p;
  p += (bytes + 255) & ~(size_t)255;
  return r;
}

template <int NT>
static inline void launch_gemm(const half_t* A, const half_t* W, float* C,
                               int Mpad, int N, int K, hipStream_t s) {
  dim3 grid(N / (16 * NT), Mpad / 16);
  gemm_wmma_f16<NT><<<grid, 32, 0, s>>>(A, W, C, N, K);
}

static inline long nblk(long tot) { return (tot + 255) / 256; }

extern "C" void kernel_launch(void* const* d_in, const int* in_sizes, int n_in,
                              void* d_out, int out_size, void* d_ws, size_t ws_size,
                              hipStream_t stream) {
  // ---- inputs (setup_inputs dict order, params flattened) ----
  const float* x     = (const float*)d_in[0];
  const float* prev  = (const float*)d_in[1];
  const int*   lens  = (const int*)  d_in[2];
  const float* ln_g  = (const float*)d_in[3];
  const float* ln_b  = (const float*)d_in[4];
  // temporal mamba params
  const float* t_in   = (const float*)d_in[5];
  const float* t_cw   = (const float*)d_in[6];
  const float* t_cb   = (const float*)d_in[7];
  const float* t_xp   = (const float*)d_in[8];
  const float* t_dtw  = (const float*)d_in[9];
  const float* t_dtb  = (const float*)d_in[10];
  const float* t_alog = (const float*)d_in[11];
  const float* t_D    = (const float*)d_in[12];
  const float* t_out  = (const float*)d_in[13];
  // spatial mamba params
  const float* s_in   = (const float*)d_in[14];
  const float* s_cw   = (const float*)d_in[15];
  const float* s_cb   = (const float*)d_in[16];
  const float* s_xp   = (const float*)d_in[17];
  const float* s_dtw  = (const float*)d_in[18];
  const float* s_dtb  = (const float*)d_in[19];
  const float* s_alog = (const float*)d_in[20];
  const float* s_D    = (const float*)d_in[21];
  const float* s_out  = (const float*)d_in[22];

  float* out = (float*)d_out;

  // ---- workspace carve ----
  char* p = (char*)d_ws;
  float*  xn      = (float*) bump(p, (size_t)NP_PAD * EMB * 4);
  half_t* comb_h  = (half_t*)bump(p, (size_t)MT_PAD * EMB * 2);   // also ys_h (spatial)
  float*  xz      = (float*) bump(p, (size_t)MT_PAD * DIN2 * 4);
  float*  xc_f    = (float*) bump(p, (size_t)MT_PAD * DIN * 4);
  half_t* xc_h    = (half_t*)bump(p, (size_t)MT_PAD * DIN * 2);
  float*  xdbl    = (float*) bump(p, (size_t)MT_PAD * XPN * 4);
  half_t* dtin_h  = (half_t*)bump(p, (size_t)MT_PAD * KDT * 2);
  float*  dtraw   = (float*) bump(p, (size_t)MT_PAD * DIN * 4);
  half_t* yfin_h  = (half_t*)bump(p, (size_t)NP_PAD * DIN * 2);   // also gathered_h
  float*  temp_f  = (float*) bump(p, (size_t)NP_PAD * EMB * 4);
  float*  spa_f   = (float*) bump(p, (size_t)MS * EMB * 4);
  float*  Aneg_t  = (float*) bump(p, (size_t)DIN * DSTATE * 4);
  float*  Aneg_s  = (float*) bump(p, (size_t)DIN * DSTATE * 4);
  half_t* w_in_t  = (half_t*)bump(p, (size_t)DIN2 * EMB * 2);
  half_t* w_in_s  = (half_t*)bump(p, (size_t)DIN2 * EMB * 2);
  half_t* w_xp_t  = (half_t*)bump(p, (size_t)XPN * DIN * 2);
  half_t* w_xp_s  = (half_t*)bump(p, (size_t)XPN * DIN * 2);
  half_t* w_dt_t  = (half_t*)bump(p, (size_t)DIN * KDT * 2);
  half_t* w_dt_s  = (half_t*)bump(p, (size_t)DIN * KDT * 2);
  half_t* w_out_t = (half_t*)bump(p, (size_t)EMB * DIN * 2);
  half_t* w_out_s = (half_t*)bump(p, (size_t)EMB * DIN * 2);
  int*    off     = (int*)   bump(p, 256);
  half_t* ys_h      = comb_h;   // spatial scan output (aliases combined)
  half_t* gather_h  = yfin_h;   // spatial gathered input (aliases yfin)

  // ---- weight preparation (f16 + padding, A = -exp(A_log)) ----
  k_convert_w<<<nblk((long)DIN2*EMB), 256, 0, stream>>>(w_in_t,  t_in,  DIN2, EMB, DIN2, EMB);
  k_convert_w<<<nblk((long)DIN2*EMB), 256, 0, stream>>>(w_in_s,  s_in,  DIN2, EMB, DIN2, EMB);
  k_convert_w<<<nblk((long)XPN*DIN),  256, 0, stream>>>(w_xp_t,  t_xp,  DTRANK+2*DSTATE, DIN, XPN, DIN);
  k_convert_w<<<nblk((long)XPN*DIN),  256, 0, stream>>>(w_xp_s,  s_xp,  DTRANK+2*DSTATE, DIN, XPN, DIN);
  k_convert_w<<<nblk((long)DIN*KDT),  256, 0, stream>>>(w_dt_t,  t_dtw, DIN, DTRANK, DIN, KDT);
  k_convert_w<<<nblk((long)DIN*KDT),  256, 0, stream>>>(w_dt_s,  s_dtw, DIN, DTRANK, DIN, KDT);
  k_convert_w<<<nblk((long)EMB*DIN),  256, 0, stream>>>(w_out_t, t_out, EMB, DIN, EMB, DIN);
  k_convert_w<<<nblk((long)EMB*DIN),  256, 0, stream>>>(w_out_s, s_out, EMB, DIN, EMB, DIN);
  k_prep_A<<<nblk(DIN*DSTATE), 256, 0, stream>>>(Aneg_t, t_alog);
  k_prep_A<<<nblk(DIN*DSTATE), 256, 0, stream>>>(Aneg_s, s_alog);

  // ---- LayerNorm ----
  k_layernorm<<<NPEDS, 256, 0, stream>>>(xn, x, ln_g, ln_b);

  // ================= temporal mamba (B=2347, L=12) =================
  k_build_combined<<<nblk((long)MT*EMB), 256, 0, stream>>>(comb_h, prev, xn);
  launch_gemm<4>(comb_h, w_in_t, xz, MT_PAD, DIN2, EMB, stream);
  k_conv_silu<<<nblk((long)MT*DIN), 256, 0, stream>>>(xc_f, xc_h, xz, t_cw, t_cb, NPEDS, TSEQ);
  launch_gemm<3>(xc_h, w_xp_t, xdbl, MT_PAD, XPN, DIN, stream);
  k_prep_dt<<<nblk((long)MT*KDT), 256, 0, stream>>>(dtin_h, xdbl, (long)MT);
  launch_gemm<4>(dtin_h, w_dt_t, dtraw, MT_PAD, DIN, KDT, stream);
  k_scan<<<nblk((long)NPEDS*DIN), 256, 0, stream>>>(
      yfin_h, xz, xc_f, xdbl, dtraw, Aneg_t, t_D, t_dtb, NPEDS, TSEQ, /*last_only=*/1);
  launch_gemm<4>(yfin_h, w_out_t, temp_f, NP_PAD, EMB, DIN, stream);

  // ================= spatial mamba (B=64, L=58) =================
  k_offsets<<<1, 32, 0, stream>>>(off, lens);
  k_gather<<<nblk((long)MS*EMB), 256, 0, stream>>>(gather_h, xn, off, lens);
  launch_gemm<4>(gather_h, w_in_s, xz, MS, DIN2, EMB, stream);
  k_conv_silu<<<nblk((long)MS*DIN), 256, 0, stream>>>(xc_f, xc_h, xz, s_cw, s_cb, NSCENES, MAXP);
  launch_gemm<3>(xc_h, w_xp_s, xdbl, MS, XPN, DIN, stream);
  k_prep_dt<<<nblk((long)MS*KDT), 256, 0, stream>>>(dtin_h, xdbl, (long)MS);
  launch_gemm<4>(dtin_h, w_dt_s, dtraw, MS, DIN, KDT, stream);
  k_scan<<<nblk((long)NSCENES*DIN), 256, 0, stream>>>(
      ys_h, xz, xc_f, xdbl, dtraw, Aneg_s, s_D, s_dtb, NSCENES, MAXP, /*last_only=*/0);
  launch_gemm<4>(ys_h, w_out_s, spa_f, MS, EMB, DIN, stream);

  // ---- residual combine ----
  k_final<<<nblk((long)MS*EMB), 256, 0, stream>>>(out, xn, temp_f, spa_f, off, lens);

  (void)in_sizes; (void)n_in; (void)out_size; (void)ws_size;
}
